// BottleneckSystolic_50414326121235
// MI455X (gfx1250) — compile-verified, matrix operand order checked by reading, not run on hardware
//
#include <hip/hip_runtime.h>
#include <hip/hip_bf16.h>

typedef __bf16 bf16_t;
typedef __attribute__((ext_vector_type(8)))  bf16_t v8bf;
typedef __attribute__((ext_vector_type(16))) bf16_t v16bf;
typedef __attribute__((ext_vector_type(8)))  float  v8f;

// ---------------------------------------------------------------------------
// Workspace layout (bytes)
// ---------------------------------------------------------------------------
#define OFF_W1  (size_t)0                          // 256*1024 bf16
#define OFF_W2  (size_t)(524288)                   // 256*9*256 bf16 ([co][tap][ci])
#define OFF_W3  (size_t)(1703936)                  // 1024*256 bf16
#define OFF_S1  (size_t)(2228224)
#define OFF_O1  (OFF_S1 + 1024)
#define OFF_S2  (OFF_O1 + 1024)
#define OFF_O2  (OFF_S2 + 1024)
#define OFF_S3  (OFF_O2 + 1024)
#define OFF_O3  (OFF_S3 + 4096)
#define OFF_T1  (size_t)(2240512)                  // 64*256*784 bf16
#define OFF_T2  (OFF_T1 + (size_t)25690112)

// ---------------------------------------------------------------------------
// WMMA fragment loader (LDS tiles with row stride 32 bf16, or global rows).
// ISA 16-bit A-matrix 16x32 layout: lanes 0-15 hold K {0..7,16..23},
// lanes 16-31 hold K {8..15,24..31}: two 16B chunks at +hi*16B, +hi*16B+32B.
// ---------------------------------------------------------------------------
__device__ __forceinline__ v16bf frag_ld(const bf16_t* p) {
    const unsigned hi = (threadIdx.x >> 4) & 1u;
    const bf16_t* q = p + hi * 8;
    v8bf lo = *reinterpret_cast<const v8bf*>(q);
    v8bf hb = *reinterpret_cast<const v8bf*>(q + 16);
    return __builtin_shufflevector(lo, hb, 0,1,2,3,4,5,6,7,8,9,10,11,12,13,14,15);
}
__device__ __forceinline__ v16bf frag_ld_rows(const bf16_t* base) {
    return frag_ld(base + (threadIdx.x & 15) * 32);
}
__device__ __forceinline__ v8f wmma_bf16(v16bf a, v16bf b, v8f c) {
    return __builtin_amdgcn_wmma_f32_16x16x32_bf16(false, a, false, b, (short)0, c, false, false);
}

// ---------------------------------------------------------------------------
// Prep kernels
// ---------------------------------------------------------------------------
__global__ void k_cvt(const float* __restrict__ s, bf16_t* __restrict__ d, int n) {
    int i = blockIdx.x * blockDim.x + threadIdx.x;
    if (i < n) d[i] = (bf16_t)s[i];
}
__global__ void k_cvt_w2(const float* __restrict__ s, bf16_t* __restrict__ d) {
    int i = blockIdx.x * blockDim.x + threadIdx.x;
    if (i < 256 * 9 * 256) {
        int co = i / 2304, r = i % 2304, tap = r / 256, ci = r % 256;
        d[i] = (bf16_t)s[(co * 256 + ci) * 9 + tap];
    }
}
__global__ void k_bn(const float* __restrict__ g, const float* __restrict__ be,
                     const float* __restrict__ mm, const float* __restrict__ vv,
                     float* __restrict__ sc, float* __restrict__ of, int n) {
    int i = blockIdx.x * blockDim.x + threadIdx.x;
    if (i < n) {
        float s = g[i] * __frsqrt_rn(vv[i] + 1e-5f);
        sc[i] = s;
        of[i] = be[i] - mm[i] * s;
    }
}

// ---------------------------------------------------------------------------
// K1: 1x1 conv (1024->256) + BN + ReLU.
// grid (2 m-tiles, 7 n-tiles, 64 batch), 256 thr = 8 waves, 2 waves/SIMD.
// ---------------------------------------------------------------------------
__global__ __launch_bounds__(256)
__attribute__((amdgpu_waves_per_eu(2)))
void k_conv1(
    const float* __restrict__ x, const bf16_t* __restrict__ w1,
    const float* __restrict__ sc, const float* __restrict__ of,
    bf16_t* __restrict__ t1)
{
    __shared__ bf16_t Blds[112 * 32];  // [px][k] transposed, 7 KB
    const int tid = threadIdx.x, lane = tid & 31, wv = tid >> 5;
    const int m0 = blockIdx.x * 128;
    const int p0 = blockIdx.y * 112;
    const size_t b = blockIdx.z;
    const size_t arow = (size_t)(m0 + wv * 16 + (lane & 15)) * 1024;

    float2 breg[7];
    auto load_b = [&](int k0) {
#pragma unroll
        for (int u = 0; u < 7; ++u) {
            int idx = tid + u * 256;
            int k = idx / 56, n2 = idx % 56;
            breg[u] = *reinterpret_cast<const float2*>(
                x + (b * 1024 + k0 + k) * 784 + p0 + 2 * n2);
        }
    };

    v8f acc[7];
    for (int j = 0; j < 7; ++j) acc[j] = v8f{0.f,0.f,0.f,0.f,0.f,0.f,0.f,0.f};

    load_b(0);
    for (int k0 = 0; k0 < 1024; k0 += 32) {
        v16bf a = frag_ld(w1 + arow + k0);   // issue early; L2 latency hidden
        __syncthreads();
#pragma unroll
        for (int u = 0; u < 7; ++u) {
            int idx = tid + u * 256;
            int k = idx / 56, n2 = idx % 56;
            Blds[(2 * n2) * 32 + k]     = (bf16_t)breg[u].x;
            Blds[(2 * n2 + 1) * 32 + k] = (bf16_t)breg[u].y;
        }
        __syncthreads();
        if (k0 + 32 < 1024) load_b(k0 + 32);   // overlap next chunk with WMMAs
        v16bf bb[7];
#pragma unroll
        for (int j = 0; j < 7; ++j) bb[j] = frag_ld_rows(Blds + j * 16 * 32);
        __builtin_amdgcn_sched_barrier(0);     // keep all 7 frags live: batched
#pragma unroll                                 // ds_loads, partial dscnt waits
        for (int j = 0; j < 7; ++j) acc[j] = wmma_bf16(a, bb[j], acc[j]);
    }
    const int hi = lane >> 4, col = lane & 15;
#pragma unroll
    for (int j = 0; j < 7; ++j) {
        int p = p0 + j * 16 + col;
#pragma unroll
        for (int i = 0; i < 8; ++i) {
            int co = m0 + wv * 16 + i + 8 * hi;
            float v = acc[j][i] * sc[co] + of[co];
            v = v > 0.f ? v : 0.f;
            t1[(b * 256 + co) * 784 + p] = (bf16_t)v;
        }
    }
}

// ---------------------------------------------------------------------------
// K2: 3x3 conv + BN + ReLU, implicit GEMM. Spatial tile 4x28 = 112 px,
// halo tile 6x30x32 in LDS (pixel-major). All 9 tap A-fragments preloaded
// from global before the (barrier-free) tap loop: 63 WMMAs per K-chunk.
// ---------------------------------------------------------------------------
__global__ __launch_bounds__(256)
__attribute__((amdgpu_waves_per_eu(2)))
void k_conv2(
    const bf16_t* __restrict__ t1, const bf16_t* __restrict__ w2t,
    const float* __restrict__ sc, const float* __restrict__ of,
    bf16_t* __restrict__ t2)
{
    __shared__ bf16_t In[6 * 30 * 32];   // [row][col][k] 11.25 KB
    const int tid = threadIdx.x, lane = tid & 31, wv = tid >> 5;
    const int m0 = blockIdx.x * 128;
    const int r0 = blockIdx.y * 4;
    const size_t b = blockIdx.z;
    const size_t arow = (size_t)(m0 + wv * 16 + (lane & 15)) * 9;

    bf16_t ireg[23];
    auto load_in = [&](int c0) {
#pragma unroll
        for (int u = 0; u < 23; ++u) {
            int idx = tid + u * 256;
            bf16_t v = (bf16_t)0.f;
            if (idx < 5760) {
                int k = idx / 180, rc = idx % 180;
                int rr = rc / 30, cc = rc % 30;
                int gr = r0 - 1 + rr, gc = cc - 1;
                if (gr >= 0 && gr < 28 && gc >= 0 && gc < 28)
                    v = t1[(b * 256 + c0 + k) * 784 + gr * 28 + gc];
            }
            ireg[u] = v;
        }
    };

    v8f acc[7];
    for (int j = 0; j < 7; ++j) acc[j] = v8f{0.f,0.f,0.f,0.f,0.f,0.f,0.f,0.f};

    load_in(0);
    for (int c0 = 0; c0 < 256; c0 += 32) {
        // preload all 9 tap A-fragments for this K-chunk (L2-resident weights)
        v16bf aa[9];
#pragma unroll
        for (int tap = 0; tap < 9; ++tap)
            aa[tap] = frag_ld(w2t + (arow + tap) * 256 + c0);
        __syncthreads();
#pragma unroll
        for (int u = 0; u < 23; ++u) {
            int idx = tid + u * 256;
            if (idx < 5760) In[(idx % 180) * 32 + (idx / 180)] = ireg[u];
        }
        __syncthreads();
        if (c0 + 32 < 256) load_in(c0 + 32);
#pragma unroll
        for (int tap = 0; tap < 9; ++tap) {
            const int kh = tap / 3, kw = tap % 3;
            v16bf bb[7];
#pragma unroll
            for (int j = 0; j < 7; ++j) {
                int pix = j * 16 + (lane & 15);
                int rr = pix / 28, c = pix % 28;
                bb[j] = frag_ld(In + ((rr + kh) * 30 + (c + kw)) * 32);
            }
            __builtin_amdgcn_sched_barrier(0);
#pragma unroll
            for (int j = 0; j < 7; ++j) acc[j] = wmma_bf16(aa[tap], bb[j], acc[j]);
        }
    }
    const int hi = lane >> 4, col = lane & 15;
#pragma unroll
    for (int j = 0; j < 7; ++j) {
        int pix = j * 16 + col;
        int p = (r0 + pix / 28) * 28 + pix % 28;
#pragma unroll
        for (int i = 0; i < 8; ++i) {
            int co = m0 + wv * 16 + i + 8 * hi;
            float v = acc[j][i] * sc[co] + of[co];
            v = v > 0.f ? v : 0.f;
            t2[(b * 256 + co) * 784 + p] = (bf16_t)v;
        }
    }
}

// ---------------------------------------------------------------------------
// K3: 1x1 conv (256->1024) + BN + fp32 residual + ReLU -> fp32 out.
// grid (8 m-tiles, 7 n-tiles, 64 batch).
// ---------------------------------------------------------------------------
__global__ __launch_bounds__(256)
__attribute__((amdgpu_waves_per_eu(2)))
void k_conv3(
    const bf16_t* __restrict__ t2, const bf16_t* __restrict__ w3,
    const float* __restrict__ sc, const float* __restrict__ of,
    const float* __restrict__ x, float* __restrict__ out)
{
    __shared__ bf16_t Blds[112 * 32];
    const int tid = threadIdx.x, lane = tid & 31, wv = tid >> 5;
    const int m0 = blockIdx.x * 128;
    const int p0 = blockIdx.y * 112;
    const size_t b = blockIdx.z;
    const size_t arow = (size_t)(m0 + wv * 16 + (lane & 15)) * 256;

    unsigned breg[7];   // packed bf16 pair along n
    auto load_b = [&](int k0) {
#pragma unroll
        for (int u = 0; u < 7; ++u) {
            int idx = tid + u * 256;
            int k = idx / 56, n2 = idx % 56;
            breg[u] = *reinterpret_cast<const unsigned*>(
                t2 + (b * 256 + k0 + k) * 784 + p0 + 2 * n2);
        }
    };

    v8f acc[7];
    for (int j = 0; j < 7; ++j) acc[j] = v8f{0.f,0.f,0.f,0.f,0.f,0.f,0.f,0.f};

    load_b(0);
    for (int k0 = 0; k0 < 256; k0 += 32) {
        v16bf a = frag_ld(w3 + arow + k0);   // issue early
        __syncthreads();
#pragma unroll
        for (int u = 0; u < 7; ++u) {
            int idx = tid + u * 256;
            int k = idx / 56, n2 = idx % 56;
            union { unsigned w; bf16_t h[2]; } cv; cv.w = breg[u];
            Blds[(2 * n2) * 32 + k]     = cv.h[0];
            Blds[(2 * n2 + 1) * 32 + k] = cv.h[1];
        }
        __syncthreads();
        if (k0 + 32 < 256) load_b(k0 + 32);
        v16bf bb[7];
#pragma unroll
        for (int j = 0; j < 7; ++j) bb[j] = frag_ld_rows(Blds + j * 16 * 32);
        __builtin_amdgcn_sched_barrier(0);
#pragma unroll
        for (int j = 0; j < 7; ++j) acc[j] = wmma_bf16(a, bb[j], acc[j]);
    }
    const int hi = lane >> 4, col = lane & 15;
#pragma unroll
    for (int j = 0; j < 7; ++j) {
        int p = p0 + j * 16 + col;
#pragma unroll
        for (int i = 0; i < 8; ++i) {
            int co = m0 + wv * 16 + i + 8 * hi;
            size_t oidx = (b * 1024 + co) * 784 + p;
            float v = acc[j][i] * sc[co] + of[co] + x[oidx];
            out[oidx] = v > 0.f ? v : 0.f;
        }
    }
}

// ---------------------------------------------------------------------------
extern "C" void kernel_launch(void* const* d_in, const int* in_sizes, int n_in,
                              void* d_out, int out_size, void* d_ws, size_t ws_size,
                              hipStream_t stream) {
    const float* x  = (const float*)d_in[0];
    const float* w1 = (const float*)d_in[1];
    const float* w2 = (const float*)d_in[2];
    const float* w3 = (const float*)d_in[3];
    const float* g1 = (const float*)d_in[4];  const float* b1 = (const float*)d_in[5];
    const float* m1 = (const float*)d_in[6];  const float* v1 = (const float*)d_in[7];
    const float* g2 = (const float*)d_in[8];  const float* b2 = (const float*)d_in[9];
    const float* m2 = (const float*)d_in[10]; const float* v2 = (const float*)d_in[11];
    const float* g3 = (const float*)d_in[12]; const float* b3 = (const float*)d_in[13];
    const float* m3 = (const float*)d_in[14]; const float* v3 = (const float*)d_in[15];
    float* out = (float*)d_out;
    char* ws = (char*)d_ws;

    bf16_t* w1b = (bf16_t*)(ws + OFF_W1);
    bf16_t* w2b = (bf16_t*)(ws + OFF_W2);
    bf16_t* w3b = (bf16_t*)(ws + OFF_W3);
    float*  s1  = (float*)(ws + OFF_S1); float* o1 = (float*)(ws + OFF_O1);
    float*  s2  = (float*)(ws + OFF_S2); float* o2 = (float*)(ws + OFF_O2);
    float*  s3  = (float*)(ws + OFF_S3); float* o3 = (float*)(ws + OFF_O3);
    bf16_t* t1  = (bf16_t*)(ws + OFF_T1);
    bf16_t* t2  = (bf16_t*)(ws + OFF_T2);

    k_cvt<<<(256 * 1024 + 255) / 256, 256, 0, stream>>>(w1, w1b, 256 * 1024);
    k_cvt_w2<<<(256 * 9 * 256 + 255) / 256, 256, 0, stream>>>(w2, w2b);
    k_cvt<<<(1024 * 256 + 255) / 256, 256, 0, stream>>>(w3, w3b, 1024 * 256);
    k_bn<<<1, 256, 0, stream>>>(g1, b1, m1, v1, s1, o1, 256);
    k_bn<<<1, 256, 0, stream>>>(g2, b2, m2, v2, s2, o2, 256);
    k_bn<<<4, 256, 0, stream>>>(g3, b3, m3, v3, s3, o3, 1024);

    k_conv1<<<dim3(2, 7, 64), 256, 0, stream>>>(x, w1b, s1, o1, t1);
    k_conv2<<<dim3(2, 7, 64), 256, 0, stream>>>(t1, w2b, s2, o2, t2);
    k_conv3<<<dim3(8, 7, 64), 256, 0, stream>>>(t2, w3b, s3, o3, x, out);
}
